// CRET_sino_27350351741219
// MI455X (gfx1250) — matrix operand order repaired and checked by reading, not compiled, not run on hardware
//
#include <hip/hip_runtime.h>

typedef __attribute__((ext_vector_type(16))) __bf16 v16bf;
typedef __attribute__((ext_vector_type(8)))  __bf16 v8bf;
typedef __attribute__((ext_vector_type(8)))  float  v8f;

#define ENCC 64
#define VDIM 360
#define DDIM 512
#define DPAD 514          // 1 zero column each side
#define NCH  9
#define P2   4096
#define OMEGA 1.5707963267948966f
#define KSTEPS 18         // 9 taps * 2 K32 chunks over 64 channels
#define VCHUNKS 24        // 360/24 = 15 v-iterations per decode thread
#define VPT (VDIM / VCHUNKS)

// ---------------------------------------------------------------------------
// Kernel 1: repack conv weights [9][64][3][3] f32 -> bf16 WMMA B-fragments.
// K ordering: step s = tap*2 + kc; local K (0..31) <-> ci = 32*kc + k_local.
// B-fragment element e of lane L: k_local = (e&7) + 8*hi + 16*(e>>3), N = L&15 (9 used).
// ---------------------------------------------------------------------------
__global__ void pack_w(const float* __restrict__ w, __bf16* __restrict__ wpack) {
    int t = blockIdx.x * blockDim.x + threadIdx.x;
    if (t >= KSTEPS * 512) return;
    int s  = t >> 9;
    int L  = (t >> 4) & 31;
    int e  = t & 15;
    int hi = (L >> 4) & 1;
    int tap = s >> 1;                       // ky*3+kx
    int kc  = s & 1;
    int ci  = (kc << 5) + (e & 7) + (hi << 3) + ((e >> 3) << 4);    // 0..63
    int n   = L & 15;
    float val = (n < NCH) ? w[(n * ENCC + ci) * 9 + tap] : 0.0f;
    wpack[t] = (__bf16)val;
}

// ---------------------------------------------------------------------------
// Kernel 2: transpose+convert feat [b][ci][v][d] f32 -> featT [b][v][dp][ci] bf16
// dp = d+1; columns dp=0 and dp=513 are zero padding (kills all border preds).
// One block per (b, v, 64-wide d chunk); LDS tile for the transpose.
// ---------------------------------------------------------------------------
__global__ __launch_bounds__(256) void transpose_feat(
    const float* __restrict__ feat, __bf16* __restrict__ featT) {
    __shared__ __bf16 lds[64][72];          // [d_local][ci], row padded to 144B
    const int blk  = blockIdx.x;            // 4*360*8
    const int dblk = blk & 7;
    const int bv   = blk >> 3;              // b*360+v
    const int b    = bv / VDIM;
    const int v    = bv - b * VDIM;
    const int t    = threadIdx.x;

    const int ci = t >> 2;                  // 0..63
    const int dg = (t & 3) << 4;            // 0,16,32,48
    const float* src = feat + (((size_t)(b * ENCC + ci) * VDIM + v) * DDIM)
                            + (dblk << 6) + dg;
#pragma unroll
    for (int q = 0; q < 4; ++q) {
        float4 f = *(const float4*)(src + (q << 2));
        lds[dg + (q << 2) + 0][ci] = (__bf16)f.x;
        lds[dg + (q << 2) + 1][ci] = (__bf16)f.y;
        lds[dg + (q << 2) + 2][ci] = (__bf16)f.z;
        lds[dg + (q << 2) + 3][ci] = (__bf16)f.w;
    }
    __syncthreads();

    const int dl  = t >> 2;                 // 0..63
    const int cig = (t & 3) << 4;           // 0,16,32,48
    __bf16* dst = featT + ((size_t)bv * DPAD + (dblk << 6) + dl + 1) * ENCC + cig;
    *(v8bf*)dst       = *(const v8bf*)&lds[dl][cig];
    *(v8bf*)(dst + 8) = *(const v8bf*)&lds[dl][cig + 8];

    // zero the padding columns dp=0 (dblk 0) and dp=513 (dblk 7)
    if (t < 4) {
        v8bf z = {};
        if (dblk == 0) {
            __bf16* pz = featT + ((size_t)bv * DPAD + 0) * ENCC + (t << 4);
            *(v8bf*)pz = z; *(v8bf*)(pz + 8) = z;
        }
        if (dblk == 7) {
            __bf16* pz = featT + ((size_t)bv * DPAD + DPAD - 1) * ENCC + (t << 4);
            *(v8bf*)pz = z; *(v8bf*)(pz + 8) = z;
        }
    }
}

// ---------------------------------------------------------------------------
// Kernel 3: 3x3 conv as 9 shifted GEMMs via v_wmma_f32_16x16x32_bf16.
// One wave -> 16(spatial d) x 16(channel, 9 used) tile; A = 2x b128 per step.
// Output stored channel-interleaved [B][V][D][16] (64B = 1 cacheline/sample).
// ---------------------------------------------------------------------------
__global__ __launch_bounds__(256) void conv_wmma(
    const __bf16* __restrict__ featT, const float* __restrict__ bias,
    const __bf16* __restrict__ wpack, float* __restrict__ sino) {
    const int lane = threadIdx.x & 31;
    const int wid  = (blockIdx.x * 256 + threadIdx.x) >> 5;   // 0..46079
    const int b  = wid / (VDIM * 32);
    const int rv = wid - b * (VDIM * 32);
    const int v  = rv >> 5;
    const int d0 = (rv & 31) << 4;

    const int m  = lane & 15;
    const int hi = (lane >> 4) & 1;

    const size_t rowstride = (size_t)DPAD * ENCC;   // bf16 elems per (b,v) row
    const __bf16* base = featT + (size_t)(b * VDIM) * rowstride
                               + (size_t)(d0 + m) * ENCC;
    v8f acc = {};
    union AF { v16bf v; v8bf h[2]; };

#pragma unroll
    for (int ky = 0; ky < 3; ++ky) {
        const int vy = v + ky - 1;
        if ((unsigned)vy < (unsigned)VDIM) {        // wave-uniform branch
            const __bf16* rb = base + (size_t)vy * rowstride;
#pragma unroll
            for (int kx = 0; kx < 3; ++kx) {
#pragma unroll
                for (int kc = 0; kc < 2; ++kc) {
                    const int s = (ky * 3 + kx) * 2 + kc;
                    v16bf bfrag = *(const v16bf*)(wpack + (s << 9) + (lane << 4));
                    const __bf16* p = rb + kx * ENCC + (kc << 5) + (hi << 3);
                    AF a;
                    a.h[0] = *(const v8bf*)p;          // e 0..7 : ci +0..7
                    a.h[1] = *(const v8bf*)(p + 16);   // e 8..15: ci +16..23
                    acc = __builtin_amdgcn_wmma_f32_16x16x32_bf16(
                        false, a.v, false, bfrag, (short)0, acc, false, false);
                }
            }
        }
    }

    // C/D layout: lane<16: (M=r, N=lane); lane>=16: (M=r+8, N=lane-16)
    const int n = lane & 15;
    const float bn = (n < NCH) ? bias[n] : 0.0f;
    float* obase = sino + (((size_t)b * VDIM + v) * DDIM + d0) * 16 + n;
#pragma unroll
    for (int r = 0; r < 8; ++r) {
        int mm = r + (hi << 3);
        obase[(size_t)mm * 16] = acc[r] + bn;   // pad channels hold junk (unread)
    }
}

// ---------------------------------------------------------------------------
// Kernel 4: Fourier decode + cross-weighted shift ensemble + bilinear BP.
// Thread = (b, vchunk, p); 15 V iterations each; writes one partial.
// ---------------------------------------------------------------------------
__global__ __launch_bounds__(256) void decode(
    const float* __restrict__ sinogram, const float* __restrict__ grid,
    const float* __restrict__ sqv, const float* __restrict__ sino,
    float* __restrict__ partial) {
    int idx = blockIdx.x * 256 + threadIdx.x;   // 4*24*4096 total
    int b   = idx / (VCHUNKS * P2);
    int rem = idx - b * (VCHUNKS * P2);
    int vc  = rem >> 12;
    int p   = rem & (P2 - 1);

    const float shift = 1.0f / (float)DDIM;
    const float* gb     = grid + (size_t)b * VDIM * P2 * 2;
    const float* sb     = sqv  + (size_t)b * VDIM * P2;
    const float* sino_b = sino + (size_t)b * VDIM * DDIM * 16;
    const float* raw    = sinogram + (size_t)b * VDIM * DDIM;

    float acc = 0.0f;
    const int v0 = vc * VPT;
    for (int v = v0; v < v0 + VPT; ++v) {
        float2 g  = *(const float2*)(gb + ((size_t)v * P2 + p) * 2);
        float  sq = sb[(size_t)v * P2 + p] * 10000.0f;

        float q[2], len[2];
#pragma unroll
        for (int t = 0; t < 2; ++t) {
            float sx = (t == 0) ? -shift : shift;
            float xg = g.x + sx + 1e-6f;
            float yg = g.y + 1e-6f;
            float xu = ((xg + 1.0f) * (float)DDIM - 1.0f) * 0.5f;
            float yu = ((yg + 1.0f) * (float)VDIM - 1.0f) * 0.5f;
            int xi = (int)fminf(fmaxf(rintf(xu), 0.0f), (float)(DDIM - 1));
            int yi = (int)fminf(fmaxf(rintf(yu), 0.0f), (float)(VDIM - 1));

            const float* qp = sino_b + ((size_t)yi * DDIM + xi) * 16;
            float4 q03 = *(const float4*)qp;        // [const, cos1, cos2, cos3]
            float4 q47 = *(const float4*)(qp + 4);  // [cos4, sin1, sin2, sin3]
            float  q8  = qp[8];                     // sin4

            float qc  = ((float)xi + 0.5f) * (2.0f / (float)DDIM) - 1.0f;
            float rel = (g.x - qc) * (float)DDIM;   // uses unshifted grid x
            float s1, c1;
            __sincosf(rel * OMEGA, &s1, &c1);
            float c2 = 2.0f * c1 * c1 - 1.0f, s2 = 2.0f * s1 * c1;
            float c3 = c2 * c1 - s2 * s1,     s3 = s2 * c1 + c2 * s1;
            float c4 = 2.0f * c2 * c2 - 1.0f, s4 = 2.0f * s2 * c2;
            q[t] = q03.x + q03.y * c1 + q03.z * c2 + q03.w * c3 + q47.x * c4
                 + q47.y * s1 + q47.z * s2 + q47.w * s3 + q8 * s4;
            len[t] = fabsf(rel) + 1e-4f;
        }
        float tot = len[0] + len[1];
        acc += ((q[0] * len[1] + q[1] * len[0]) / tot) * sq;

        // bilinear back-projection of raw sinogram (border-clamped)
        float xu = fminf(fmaxf(((g.x + 1.0f) * (float)DDIM - 1.0f) * 0.5f, 0.0f),
                         (float)(DDIM - 1));
        float yu = fminf(fmaxf(((g.y + 1.0f) * (float)VDIM - 1.0f) * 0.5f, 0.0f),
                         (float)(VDIM - 1));
        float x0f = floorf(xu), y0f = floorf(yu);
        float wx = xu - x0f, wy = yu - y0f;
        int x0 = (int)x0f, y0 = (int)y0f;
        int x1 = min(x0 + 1, DDIM - 1), y1 = min(y0 + 1, VDIM - 1);
        float v00 = raw[(size_t)y0 * DDIM + x0];
        float v01 = raw[(size_t)y0 * DDIM + x1];
        float v10 = raw[(size_t)y1 * DDIM + x0];
        float v11 = raw[(size_t)y1 * DDIM + x1];
        float bp = v00 * (1.0f - wx) * (1.0f - wy) + v01 * wx * (1.0f - wy)
                 + v10 * (1.0f - wx) * wy + v11 * wx * wy;
        acc += bp * sq;
    }
    partial[idx] = acc;
}

// ---------------------------------------------------------------------------
// Kernel 5: deterministic fixed-order reduction of the 24 v-chunk partials.
// ---------------------------------------------------------------------------
__global__ void reduce_out(const float* __restrict__ partial,
                           float* __restrict__ out) {
    int i = blockIdx.x * 256 + threadIdx.x;   // 16384
    int p = i & (P2 - 1);
    int b = i >> 12;
    float s = 0.0f;
#pragma unroll
    for (int vc = 0; vc < VCHUNKS; ++vc)
        s += partial[((size_t)(b * VCHUNKS + vc)) * P2 + p];
    out[i] = s;
}

extern "C" void kernel_launch(void* const* d_in, const int* in_sizes, int n_in,
                              void* d_out, int out_size, void* d_ws, size_t ws_size,
                              hipStream_t stream) {
    const float* sinogram = (const float*)d_in[0];   // [4,1,360,512]
    const float* feat     = (const float*)d_in[1];   // [4,64,360,512]
    const float* grid     = (const float*)d_in[2];   // [4,360,4096,2]
    const float* sqv      = (const float*)d_in[3];   // [4,1,360,4096]
    const float* adapt_w  = (const float*)d_in[4];   // [9,64,3,3]
    const float* adapt_b  = (const float*)d_in[5];   // [9]
    float* out = (float*)d_out;                      // [4,1,64,64]

    char* ws = (char*)d_ws;
    const size_t featT_bytes = (size_t)4 * VDIM * DPAD * ENCC * 2;   // ~94.8 MB
    const size_t sino_bytes  = (size_t)4 * VDIM * DDIM * 16 * 4;     // ~47.2 MB
    __bf16* wpack = (__bf16*)ws;                                     // 18 KB
    __bf16* featT = (__bf16*)(ws + 32768);
    float*  sino  = (float*)(ws + 32768 + featT_bytes);
    float*  part  = (float*)(ws + 32768 + featT_bytes + sino_bytes);

    pack_w        <<<(KSTEPS * 512 + 255) / 256, 256, 0, stream>>>(adapt_w, wpack);
    transpose_feat<<<4 * VDIM * 8,               256, 0, stream>>>(feat, featT);
    conv_wmma     <<<(4 * VDIM * 32) / 8,        256, 0, stream>>>(featT, adapt_b, wpack, sino);
    decode        <<<(4 * VCHUNKS * P2) / 256,   256, 0, stream>>>(sinogram, grid, sqv, sino, part);
    reduce_out    <<<(4 * P2) / 256,             256, 0, stream>>>(part, out);
}
